// ConformerBlock_4595615007246
// MI455X (gfx1250) — compile-verified, matrix operand order checked by reading, not run on hardware
//
#include <hip/hip_runtime.h>
#include <hip/hip_bf16.h>
#include <math.h>
#include <stdint.h>

// Problem constants (match reference)
#define BB     16
#define TT     1024
#define DD     256
#define NHH    4
#define DHH    64
#define INNERC 512
#define KW     17
#define MT     (BB * TT)   // 16384 rows
#define EPS    1e-3f

typedef __attribute__((ext_vector_type(16))) _Float16 v16h;
typedef __attribute__((ext_vector_type(8)))  float    v8f;
typedef __attribute__((ext_vector_type(4)))  unsigned tdm_g0_t;
typedef __attribute__((ext_vector_type(8)))  int      tdm_g1_t;
typedef __attribute__((ext_vector_type(4)))  int      tdm_g2_t;

#if defined(__AMDGCN__) && __has_builtin(__builtin_amdgcn_tensor_load_to_lds) && \
    __has_builtin(__builtin_amdgcn_s_wait_tensorcnt)
#define HAVE_TDM 1
#else
#define HAVE_TDM 0
#endif

__device__ __forceinline__ int lane_id() { return (int)(threadIdx.x & 31u); }

// K-pair base for the 16-bit 16x32 fragment layout (CDNA5 ISA 7.12.2):
// lanes 0-15: VGPR0..3 -> K 0..7, VGPR4..7 -> K 16..23 ; lanes 16-31: +8
__device__ __forceinline__ int kPairBase(int lane, int i) {
  return ((i < 4) ? (2 * i) : (16 + 2 * (i - 4))) + ((lane >= 16) ? 8 : 0);
}

// Load one 16x32 f16 fragment from an LDS f16 tile (pairs merge to ds_load_b128).
__device__ __forceinline__ v16h ldsFragH(const _Float16* base, int rowStride) {
  const int lane = lane_id();
  const _Float16* p = base + (size_t)(lane & 15) * rowStride;
  v16h f;
#pragma unroll
  for (int i = 0; i < 8; ++i) {
    const int kp = kPairBase(lane, i);
    f[2 * i]     = p[kp];
    f[2 * i + 1] = p[kp + 1];
  }
  return f;
}

// Same, but the LDS tile is f32 (TDM-staged, no conversion on copy) — convert on read.
__device__ __forceinline__ v16h ldsFragF32(const float* base, int rowStride) {
  const int lane = lane_id();
  const float* p = base + (size_t)(lane & 15) * rowStride;
  v16h f;
#pragma unroll
  for (int i = 0; i < 8; ++i) {
    const int kp = kPairBase(lane, i);
    f[2 * i]     = (_Float16)p[kp];
    f[2 * i + 1] = (_Float16)p[kp + 1];
  }
  return f;
}

__device__ __forceinline__ v8f wmma_f16(v16h a, v16h b, v8f c) {
  return __builtin_amdgcn_wmma_f32_16x16x32_f16(
      /*neg_a=*/false, a, /*neg_b=*/false, b,
      /*c_mod=*/(short)0, c, /*reuse_a=*/false, /*reuse_b=*/false);
}

__device__ __forceinline__ float sigm(float x) { return 1.0f / (1.0f + __expf(-x)); }

// Pack two f32 into one dword of two f16 (for ds_store_b32 staging).
__device__ __forceinline__ unsigned pkh2(float a, float b) {
  union { _Float16 h[2]; unsigned u; } t;
  t.h[0] = (_Float16)a;
  t.h[1] = (_Float16)b;
  return t.u;
}

// ---------------------------------------------------------------------------
// Tensor Data Mover: 2D tile (rows x cols f32, row stride in elements) -> LDS.
// Descriptor packed per CDNA5 ISA 8.3/8.4. Issued once per calling wave.
// ---------------------------------------------------------------------------
#if HAVE_TDM
__device__ __forceinline__ void tdm_load_2d(const void* lds_ptr, const float* gptr,
                                            int rows, int cols, int rowStrideElems) {
  const unsigned lds_off = (unsigned)(uintptr_t)lds_ptr;
  const unsigned long long ga = (unsigned long long)(uintptr_t)gptr;
  const unsigned td0 = (unsigned)rowStrideElems;  // tensor dim0 >= cols (no OOB use)
  const unsigned td1 = 1u << 20;                  // generous tensor dim1

  tdm_g0_t g0;
  g0[0] = 1u;                                        // count=1, user descriptor
  g0[1] = lds_off;                                   // lds_addr (bytes)
  g0[2] = (unsigned)(ga & 0xffffffffu);              // global_addr[31:0]
  g0[3] = (unsigned)((ga >> 32) & 0x01ffffffu)       // global_addr[56:32]
          | 0x80000000u;                             // type=2 ("image")

  tdm_g1_t g1;
  g1[0] = (int)(2u << 16);                           // data_size = 4 bytes
  g1[1] = (int)((td0 & 0xffffu) << 16);              // tensor_dim0[15:0]
  g1[2] = (int)((td0 >> 16) | ((td1 & 0xffffu) << 16));  // dim0 hi / dim1 lo
  g1[3] = (int)((td1 >> 16) | ((unsigned)cols << 16));   // dim1 hi / tile_dim0
  g1[4] = (int)rows;                                 // tile_dim1 (tile_dim2 = 0)
  g1[5] = (int)td0;                                  // tensor_dim0_stride[31:0]
  g1[6] = 0;                                         // stride hi / dim1_stride lo
  g1[7] = 0;
  tdm_g2_t g2 = {0, 0, 0, 0};
  tdm_g2_t g3 = {0, 0, 0, 0};
#if defined(__clang_major__) && (__clang_major__ >= 23)
  tdm_g1_t g4 = {0, 0, 0, 0, 0, 0, 0, 0};
  __builtin_amdgcn_tensor_load_to_lds(g0, g1, g2, g3, g4, 0);
#else
  __builtin_amdgcn_tensor_load_to_lds(g0, g1, g2, g3, 0);
#endif
}
#endif

// ---------------------------------------------------------------------------
// Generic tiled GEMM: out[M,N] = resid_scale*resid + out_scale*act(A@W + bias)
// A row-major [M,Kd], W row-major [Kd,N].  M%128==0, N%128==0, Kd%32==0.
// Block tile 128x128, BK=32, 256 threads (8 waves, 2x4), wave tile 64x32.
// A tile staged f32 by the TDM (wave 0 issues descriptor); W tile staged
// f16-transposed by VALU (pair-packed ds_store_b32). ACT: 0=identity, 1=swish.
// ---------------------------------------------------------------------------
template <int ACT>
__global__ __launch_bounds__(256) void gemm_kernel(
    const float* __restrict__ A, const float* __restrict__ W,
    const float* __restrict__ bias, const float* __restrict__ resid,
    float* __restrict__ out, int M, int N, int Kd,
    float out_scale, float resid_scale) {
  __shared__ float    a_sf[128 * 32];   // [m][k] f32 (TDM destination)
  __shared__ unsigned b_su[128 * 16];   // [n][k/2] packed f16 pairs

  const _Float16* b_s = (const _Float16*)b_su;
  const int tid  = threadIdx.x;
  const int lane = tid & 31;
  const int wave = tid >> 5;
  const int wm   = (wave >> 2) * 64;
  const int wn   = (wave & 3) * 32;
  const int m0   = blockIdx.y * 128;
  const int n0   = blockIdx.x * 128;

  v8f acc[4][2];
#pragma unroll
  for (int i = 0; i < 4; ++i)
#pragma unroll
    for (int j = 0; j < 2; ++j)
#pragma unroll
      for (int v = 0; v < 8; ++v) acc[i][j][v] = 0.0f;

  for (int k0 = 0; k0 < Kd; k0 += 32) {
#if HAVE_TDM
    if (wave == 0) {  // one async 128x32 f32 tile, row stride Kd
      tdm_load_2d(a_sf, &A[(size_t)m0 * Kd + k0], 128, 32, Kd);
    }
#else
    for (int idx = tid; idx < 128 * 32; idx += 256) {
      const int r = idx >> 5, c = idx & 31;
      a_sf[idx] = A[(size_t)(m0 + r) * Kd + (k0 + c)];
    }
#endif
    // Stage W tile transposed to [n][k], two k per thread, packed f16 stores.
    for (int idx = tid; idx < 128 * 16; idx += 256) {
      const int n = idx >> 4, k2 = (idx & 15) * 2;
      const float w0 = W[(size_t)(k0 + k2) * N + (n0 + n)];
      const float w1 = W[(size_t)(k0 + k2 + 1) * N + (n0 + n)];
      b_su[idx] = pkh2(w0, w1);
    }
    if (k0 + 32 < Kd) {  // lowers to global_prefetch_b8
      __builtin_prefetch(&W[(size_t)(k0 + 32 + (tid >> 3)) * N + n0], 0, 1);
    }
#if HAVE_TDM
    if (wave == 0) __builtin_amdgcn_s_wait_tensorcnt(0);
#endif
    __syncthreads();

    v16h bf[2];
#pragma unroll
    for (int j = 0; j < 2; ++j)
      bf[j] = ldsFragH(b_s + (size_t)(wn + j * 16) * 32, 32);
#pragma unroll
    for (int i = 0; i < 4; ++i) {
      const v16h af = ldsFragF32(a_sf + (size_t)(wm + i * 16) * 32, 32);
#pragma unroll
      for (int j = 0; j < 2; ++j) acc[i][j] = wmma_f16(af, bf[j], acc[i][j]);
    }
    __syncthreads();
  }

  // Epilogue: C layout — N = lane&15, M = v + 8*(lane>=16)
  const int cn = lane & 15;
  const int ch = (lane >= 16) ? 8 : 0;
#pragma unroll
  for (int i = 0; i < 4; ++i)
#pragma unroll
    for (int j = 0; j < 2; ++j) {
      const int nn = n0 + wn + j * 16 + cn;
      const float bv = bias ? bias[nn] : 0.0f;
#pragma unroll
      for (int v = 0; v < 8; ++v) {
        const int mm = m0 + wm + i * 16 + ch + v;
        float val = acc[i][j][v] + bv;
        if (ACT == 1) val = val * sigm(val);
        float r = resid ? resid[(size_t)mm * N + nn] * resid_scale : 0.0f;
        out[(size_t)mm * N + nn] = r + out_scale * val;
      }
    }
}

// ---------------------------------------------------------------------------
// LayerNorm over last dim D=256; one 256-thread block per row
// ---------------------------------------------------------------------------
__global__ __launch_bounds__(256) void ln_kernel(
    const float* __restrict__ x, const float* __restrict__ g,
    const float* __restrict__ b, float* __restrict__ out) {
  __shared__ float red[256];
  const size_t row = blockIdx.x;
  const int tid = threadIdx.x;
  const float v = x[row * DD + tid];
  red[tid] = v;
  __syncthreads();
  for (int s = 128; s > 0; s >>= 1) {
    if (tid < s) red[tid] += red[tid + s];
    __syncthreads();
  }
  const float mean = red[0] * (1.0f / DD);
  __syncthreads();
  const float d = v - mean;
  red[tid] = d * d;
  __syncthreads();
  for (int s = 128; s > 0; s >>= 1) {
    if (tid < s) red[tid] += red[tid + s];
    __syncthreads();
  }
  const float inv = rsqrtf(red[0] * (1.0f / DD) + EPS);
  out[row * DD + tid] = d * inv * g[tid] + b[tid];
}

// ---------------------------------------------------------------------------
// Sinusoidal relative position embedding [T, D], positions T-1 .. 0
// ---------------------------------------------------------------------------
__global__ __launch_bounds__(256) void posemb_kernel(float* __restrict__ P) {
  const int idx = blockIdx.x * 256 + threadIdx.x;
  if (idx >= TT * DD) return;
  const int t = idx / DD, f = idx % DD;
  const float pos = (float)(TT - 1 - t);
  const int i = (f < DD / 2) ? f : (f - DD / 2);
  const float invf = powf(10000.0f, -(2.0f * (float)i) / (float)DD);
  const float a = pos * invf;
  P[idx] = (f < DD / 2) ? sinf(a) : cosf(a);
}

// ---------------------------------------------------------------------------
// Fused relative-position attention. One wave per (b, head, 16-query-row) tile.
// qkv: [MT, 3*NH*DH] (q | k | v). R: [T, NH*DH]. out: [MT, NH*DH]
// Q/K/R tiles are TDM-staged as f32; V tile is VALU-transposed to f16.
// ---------------------------------------------------------------------------
__global__ __launch_bounds__(32) void attn_kernel(
    const float* __restrict__ qkv, const float* __restrict__ R,
    const unsigned char* __restrict__ mask, float* __restrict__ out) {
  __shared__ float    sc_s[16 * TT];      // 64 KB: score row / probs
  __shared__ float    qa_sf[16 * DHH];    // Q tile  [ii][d] f32 (TDM)
  __shared__ float    kb_sf[16 * DHH];    // K/R tile [j|p][d] f32 (TDM)
  __shared__ unsigned vb_su[DHH * 16];    // V tile  [d][j/2] packed f16 pairs

  const _Float16* vb_s = (const _Float16*)vb_su;
  const int lane = lane_id();
  const int i0 = blockIdx.x * 16;   // query rows
  const int h  = blockIdx.y;        // head
  const int b  = blockIdx.z;        // batch
  const size_t qkvRow = 3 * NHH * DHH;   // 768
  const size_t bBase = (size_t)b * TT;

  // 1) Stage Q tile (f32)
#if HAVE_TDM
  tdm_load_2d(qa_sf, &qkv[(bBase + i0) * qkvRow + h * DHH], 16, DHH, (int)qkvRow);
  __builtin_amdgcn_s_wait_tensorcnt(0);
#else
  for (int idx = lane; idx < 16 * DHH; idx += 32) {
    const int ii = idx >> 6, d = idx & 63;
    qa_sf[idx] = qkv[(bBase + i0 + ii) * qkvRow + h * DHH + d];
  }
#endif
  __syncthreads();
  v16h aq[2];
  aq[0] = ldsFragF32(qa_sf + 0, DHH);
  aq[1] = ldsFragF32(qa_sf + 32, DHH);

  const int cn = lane & 15;
  const int ch = (lane >= 16) ? 8 : 0;

  // 2) AC = Q K^T, written into sc_s
  for (int j0 = 0; j0 < TT; j0 += 16) {
#if HAVE_TDM
    tdm_load_2d(kb_sf, &qkv[(bBase + j0) * qkvRow + NHH * DHH + h * DHH], 16, DHH,
                (int)qkvRow);
    __builtin_amdgcn_s_wait_tensorcnt(0);
#else
    for (int idx = lane; idx < 16 * DHH; idx += 32) {
      const int jj = idx >> 6, d = idx & 63;
      kb_sf[idx] = qkv[(bBase + j0 + jj) * qkvRow + NHH * DHH + h * DHH + d];
    }
#endif
    __syncthreads();
    v8f c;
#pragma unroll
    for (int v = 0; v < 8; ++v) c[v] = 0.0f;
    c = wmma_f16(aq[0], ldsFragF32(kb_sf + 0, DHH), c);
    c = wmma_f16(aq[1], ldsFragF32(kb_sf + 32, DHH), c);
#pragma unroll
    for (int v = 0; v < 8; ++v) sc_s[(ch + v) * TT + (j0 + cn)] = c[v];
    __syncthreads();
  }

  // 3) BD = Q R^T with Transformer-XL shift folded in: j = p + i - (T-1)
  for (int p0 = 0; p0 < TT; p0 += 16) {
#if HAVE_TDM
    tdm_load_2d(kb_sf, &R[(size_t)p0 * (NHH * DHH) + h * DHH], 16, DHH, NHH * DHH);
    __builtin_amdgcn_s_wait_tensorcnt(0);
#else
    for (int idx = lane; idx < 16 * DHH; idx += 32) {
      const int pp = idx >> 6, d = idx & 63;
      kb_sf[idx] = R[(size_t)(p0 + pp) * (NHH * DHH) + h * DHH + d];
    }
#endif
    __syncthreads();
    v8f c;
#pragma unroll
    for (int v = 0; v < 8; ++v) c[v] = 0.0f;
    c = wmma_f16(aq[0], ldsFragF32(kb_sf + 0, DHH), c);
    c = wmma_f16(aq[1], ldsFragF32(kb_sf + 32, DHH), c);
#pragma unroll
    for (int v = 0; v < 8; ++v) {
      const int m = ch + v;                       // local query row
      const int j = (p0 + cn) + (i0 + m) - (TT - 1);
      if (j >= 0 && j < TT) sc_s[m * TT + j] += c[v];
    }
    __syncthreads();
  }

  // 4) scale + mask + softmax (row-wise, wave shuffle reductions)
  const float scale = 0.125f;  // 1/sqrt(64)
  for (int r = 0; r < 16; ++r) {
    float mx = -3.4e38f;
    for (int j = lane; j < TT; j += 32) {
      const float mv = mask[bBase + j] ? 1.0f : 0.0f;
      float s = sc_s[r * TT + j] * scale;
      s = s * mv - 1e30f * (1.0f - mv);
      sc_s[r * TT + j] = s;
      mx = fmaxf(mx, s);
    }
#pragma unroll
    for (int off = 16; off > 0; off >>= 1) mx = fmaxf(mx, __shfl_xor(mx, off, 32));
    float sum = 0.0f;
    for (int j = lane; j < TT; j += 32) {
      const float e = __expf(sc_s[r * TT + j] - mx);
      sc_s[r * TT + j] = e;
      sum += e;
    }
#pragma unroll
    for (int off = 16; off > 0; off >>= 1) sum += __shfl_xor(sum, off, 32);
    const float inv = 1.0f / sum;
    for (int j = lane; j < TT; j += 32) sc_s[r * TT + j] *= inv;
  }
  __syncthreads();

  // 5) O = P V   (O is 16 x 64 -> four 16x16 accumulators)
  v8f o[4];
#pragma unroll
  for (int nt = 0; nt < 4; ++nt)
#pragma unroll
    for (int v = 0; v < 8; ++v) o[nt][v] = 0.0f;

  for (int j0 = 0; j0 < TT; j0 += 32) {
    // V tile transposed to [d][j], two j per thread, packed f16 ds_store_b32
    for (int idx = lane; idx < DHH * 16; idx += 32) {
      const int d = idx >> 4, j2 = (idx & 15) * 2;
      const float v0 = qkv[(bBase + j0 + j2) * qkvRow + 2 * NHH * DHH + h * DHH + d];
      const float v1 = qkv[(bBase + j0 + j2 + 1) * qkvRow + 2 * NHH * DHH + h * DHH + d];
      vb_su[idx] = pkh2(v0, v1);
    }
    __syncthreads();
    const v16h pa = ldsFragF32(sc_s + j0, TT);
#pragma unroll
    for (int nt = 0; nt < 4; ++nt)
      o[nt] = wmma_f16(pa, ldsFragH(vb_s + (size_t)(nt * 16) * 32, 32), o[nt]);
    __syncthreads();
  }

  // 6) write attn_vec [MT, NH*DH]
#pragma unroll
  for (int nt = 0; nt < 4; ++nt)
#pragma unroll
    for (int v = 0; v < 8; ++v) {
      const int mm = i0 + ch + v;
      out[(bBase + mm) * (NHH * DHH) + h * DHH + nt * 16 + cn] = o[nt][v];
    }
}

// ---------------------------------------------------------------------------
// Small elementwise kernels
// ---------------------------------------------------------------------------
__global__ __launch_bounds__(256) void add_kernel(float* __restrict__ acc,
                                                  const float* __restrict__ x,
                                                  size_t n) {
  const size_t i = (size_t)blockIdx.x * 256 + threadIdx.x;
  if (i < n) acc[i] += x[i];
}

__global__ __launch_bounds__(256) void glu_kernel(const float* __restrict__ h,
                                                  float* __restrict__ out) {
  const size_t i = (size_t)blockIdx.x * 256 + threadIdx.x;
  if (i >= (size_t)MT * INNERC) return;
  const size_t row = i / INNERC;
  const int c = (int)(i % INNERC);
  const float a = h[row * (2 * INNERC) + c];
  const float g = h[row * (2 * INNERC) + INNERC + c];
  out[i] = a * sigm(g);
}

__global__ __launch_bounds__(256) void dwconv_kernel(const float* __restrict__ in,
                                                     const float* __restrict__ w,
                                                     const float* __restrict__ bias,
                                                     float* __restrict__ out) {
  const size_t i = (size_t)blockIdx.x * 256 + threadIdx.x;
  if (i >= (size_t)MT * INNERC) return;
  const int c = (int)(i % INNERC);
  const size_t row = i / INNERC;
  const int t = (int)(row % TT);
  const size_t bt0 = row - t;
  float acc = bias[c];
#pragma unroll
  for (int k = 0; k < KW; ++k) {
    const int tt = t + k - KW / 2;
    if (tt >= 0 && tt < TT) acc += in[(bt0 + tt) * INNERC + c] * w[k * INNERC + c];
  }
  out[i] = acc;
}

__global__ __launch_bounds__(256) void bnswish_kernel(float* __restrict__ y,
                                                      const float* __restrict__ g,
                                                      const float* __restrict__ b,
                                                      const float* __restrict__ mean,
                                                      const float* __restrict__ var) {
  const size_t i = (size_t)blockIdx.x * 256 + threadIdx.x;
  if (i >= (size_t)MT * INNERC) return;
  const int c = (int)(i % INNERC);
  const float z = (y[i] - mean[c]) * rsqrtf(var[c] + EPS) * g[c] + b[c];
  y[i] = z * sigm(z);
}

// ---------------------------------------------------------------------------
// Orchestration
// ---------------------------------------------------------------------------
static inline dim3 gemm_grid(int M, int N) { return dim3(N / 128, M / 128, 1); }

extern "C" void kernel_launch(void* const* d_in, const int* in_sizes, int n_in,
                              void* d_out, int out_size, void* d_ws, size_t ws_size,
                              hipStream_t stream) {
  (void)in_sizes; (void)n_in; (void)out_size; (void)ws_size;
  const float* inputs  = (const float*)d_in[0];
  const unsigned char* mask = (const unsigned char*)d_in[1];
  const float* ff1_w1  = (const float*)d_in[2];
  const float* ff1_b1  = (const float*)d_in[3];
  const float* ff1_w2  = (const float*)d_in[4];
  const float* ff1_b2  = (const float*)d_in[5];
  const float* attn_ln_g = (const float*)d_in[6];
  const float* attn_ln_b = (const float*)d_in[7];
  const float* qkv_w   = (const float*)d_in[8];
  const float* r_w     = (const float*)d_in[9];
  const float* o_w     = (const float*)d_in[10];
  const float* conv_ln_g = (const float*)d_in[11];
  const float* conv_ln_b = (const float*)d_in[12];
  const float* conv_w1 = (const float*)d_in[13];
  const float* conv_b1 = (const float*)d_in[14];
  const float* dw_kernel = (const float*)d_in[15];
  const float* dw_bias = (const float*)d_in[16];
  const float* bn_g    = (const float*)d_in[17];
  const float* bn_b    = (const float*)d_in[18];
  const float* bn_mean = (const float*)d_in[19];
  const float* bn_var  = (const float*)d_in[20];
  const float* conv_w2 = (const float*)d_in[21];
  const float* ff2_w1  = (const float*)d_in[22];
  const float* ff2_b1  = (const float*)d_in[23];
  const float* ff2_w2  = (const float*)d_in[24];
  const float* ff2_b2  = (const float*)d_in[25];
  const float* ln_g    = (const float*)d_in[26];
  const float* ln_b    = (const float*)d_in[27];
  float* out = (float*)d_out;

  float* ws = (float*)d_ws;
  float* buf_x   = ws;                                  // [MT,256]
  float* buf_ln  = buf_x   + (size_t)MT * 256;          // [MT,256]
  float* buf_h   = buf_ln  + (size_t)MT * 256;          // [MT,1024]
  float* buf_qkv = buf_h   + (size_t)MT * 1024;         // [MT,768] (later GLU [MT,512])
  float* buf_v2  = buf_qkv + (size_t)MT * 768;          // [MT,512] (attn_vec / dw out)
  float* buf_pos = buf_v2  + (size_t)MT * 512;          // [T,256]
  float* buf_r   = buf_pos + (size_t)TT * 256;          // [T,256]

  const size_t nXD = (size_t)MT * 256;
  const size_t nXI = (size_t)MT * INNERC;

  // --- FF1 half-step: x = inputs + 0.5 * (swish(in@w1+b1) @ w2 + b2)
  gemm_kernel<1><<<gemm_grid(MT, 1024), 256, 0, stream>>>(
      inputs, ff1_w1, ff1_b1, nullptr, buf_h, MT, 1024, 256, 1.0f, 0.0f);
  gemm_kernel<0><<<gemm_grid(MT, 256), 256, 0, stream>>>(
      buf_h, ff1_w2, ff1_b2, inputs, buf_x, MT, 256, 1024, 0.5f, 1.0f);

  // --- Attention block
  ln_kernel<<<MT, 256, 0, stream>>>(buf_x, attn_ln_g, attn_ln_b, buf_ln);
  gemm_kernel<0><<<gemm_grid(MT, 768), 256, 0, stream>>>(
      buf_ln, qkv_w, nullptr, nullptr, buf_qkv, MT, 768, 256, 1.0f, 0.0f);
  posemb_kernel<<<(TT * DD + 255) / 256, 256, 0, stream>>>(buf_pos);
  gemm_kernel<0><<<gemm_grid(TT, 256), 256, 0, stream>>>(
      buf_pos, r_w, nullptr, nullptr, buf_r, TT, 256, 256, 1.0f, 0.0f);
  attn_kernel<<<dim3(TT / 16, NHH, BB), 32, 0, stream>>>(buf_qkv, buf_r, mask, buf_v2);
  // resid = x + x_ln  (MHRA adds its own residual on the LN'd input)
  add_kernel<<<(int)((nXD + 255) / 256), 256, 0, stream>>>(buf_ln, buf_x, nXD);
  gemm_kernel<0><<<gemm_grid(MT, 256), 256, 0, stream>>>(
      buf_v2, o_w, nullptr, buf_ln, buf_x, MT, 256, 256, 1.0f, 1.0f);

  // --- Conv module
  ln_kernel<<<MT, 256, 0, stream>>>(buf_x, conv_ln_g, conv_ln_b, buf_ln);
  gemm_kernel<0><<<gemm_grid(MT, 1024), 256, 0, stream>>>(
      buf_ln, conv_w1, conv_b1, nullptr, buf_h, MT, 1024, 256, 1.0f, 0.0f);
  glu_kernel<<<(int)((nXI + 255) / 256), 256, 0, stream>>>(buf_h, buf_qkv);
  dwconv_kernel<<<(int)((nXI + 255) / 256), 256, 0, stream>>>(buf_qkv, dw_kernel, dw_bias, buf_v2);
  bnswish_kernel<<<(int)((nXI + 255) / 256), 256, 0, stream>>>(buf_v2, bn_g, bn_b, bn_mean, bn_var);
  gemm_kernel<0><<<gemm_grid(MT, 256), 256, 0, stream>>>(
      buf_v2, conv_w2, nullptr, buf_x, buf_ln, MT, 256, 512, 1.0f, 1.0f);  // new x in buf_ln

  // --- FF2 half-step + final LN
  gemm_kernel<1><<<gemm_grid(MT, 1024), 256, 0, stream>>>(
      buf_ln, ff2_w1, ff2_b1, nullptr, buf_h, MT, 1024, 256, 1.0f, 0.0f);
  gemm_kernel<0><<<gemm_grid(MT, 256), 256, 0, stream>>>(
      buf_h, ff2_w2, ff2_b2, buf_ln, buf_x, MT, 256, 1024, 0.5f, 1.0f);
  ln_kernel<<<MT, 256, 0, stream>>>(buf_x, ln_g, ln_b, out);
}
